// HammingLoss_39986145525891
// MI455X (gfx1250) — compile-verified, matrix operand order checked by reading, not run on hardware
//
#include <hip/hip_runtime.h>

typedef __attribute__((ext_vector_type(8))) int v8i;

// Exact for a, b in [0, 27): number of differing base-3 digits (3 digits).
__device__ __forceinline__ unsigned mism3(unsigned a, unsigned b) {
    unsigned a3 = (a * 43u) >> 7;   // a / 3  (exact for a < 27)
    unsigned b3 = (b * 43u) >> 7;
    unsigned a9 = (a * 57u) >> 9;   // a / 9  (exact for a < 27)
    unsigned b9 = (b * 57u) >> 9;
    unsigned r  = (unsigned)((a  - 3u * a3) != (b  - 3u * b3));   // digit 0
    r          += (unsigned)((a3 - 3u * a9) != (b3 - 3u * b9));   // digit 1
    r          += (unsigned)(a9 != b9);                           // digit 2
    return r;
}

__global__ void hl_init_ws(unsigned* ws) {
    if (threadIdx.x == 0 && blockIdx.x == 0) ws[0] = 0u;
}

// Streaming kernel. Each wave consumes contiguous 1024-element chunks:
//  - 8 rounds of coalesced int4 (b128) loads per operand (512B/wave/round)
//  - per element: idx = (a<<5)|b, one ds_load_u8 from an 864B pairwise
//    Hamming LUT in LDS (values 0..3), packed 4-per-u32
//  - each lane's 8 packed u32 = its slice of the 16x64 u8 A tile;
//    one v_wmma_i32_16x16x64_iu8 with B = all-ones reduces all 1024 counts
//    into the i32 accumulator (row-sums replicated over 16 columns -> x16).
__global__ __launch_bounds__(256) void hl_hamming_wmma(
        const int* __restrict__ A, const int* __restrict__ B,
        unsigned* __restrict__ gsum, int n) {
    // ---- build the 27x32 pairwise Hamming LUT (864 bytes) ----
    __shared__ unsigned char lut[864];
    for (unsigned i = threadIdx.x; i < 864u; i += blockDim.x) {
        const unsigned a = i >> 5, b = i & 31u;
        lut[i] = (unsigned char)((b < 27u) ? mism3(a, b) : 0u);
    }
    __shared__ unsigned bsum;
    if (threadIdx.x == 0) bsum = 0u;
    __syncthreads();

    const unsigned tid     = blockIdx.x * blockDim.x + threadIdx.x;
    const unsigned lane    = threadIdx.x & 31u;
    const unsigned wave    = tid >> 5;
    const unsigned nwaves  = (gridDim.x * blockDim.x) >> 5;
    const unsigned nchunks = (unsigned)n >> 10;   // full 1024-element chunks

    v8i acc = {};           // i32 16x16 accumulator (8 VGPRs)
    v8i ones;
    #pragma unroll
    for (int i = 0; i < 8; ++i) ones[i] = 0x01010101;  // B = all-ones u8

    for (unsigned ch = wave; ch < nchunks; ch += nwaves) {  // wave-uniform: EXEC all-1s
        const unsigned base = ch << 10;
        v8i apack;
        #pragma unroll
        for (int j = 0; j < 8; ++j) {
            const unsigned idx = base + (((unsigned)j * 32u + lane) << 2);
            const int4 a4 = *(const int4*)(A + idx);
            const int4 b4 = *(const int4*)(B + idx);
            const unsigned m0 = lut[((unsigned)a4.x << 5) | (unsigned)b4.x];
            const unsigned m1 = lut[((unsigned)a4.y << 5) | (unsigned)b4.y];
            const unsigned m2 = lut[((unsigned)a4.z << 5) | (unsigned)b4.z];
            const unsigned m3 = lut[((unsigned)a4.w << 5) | (unsigned)b4.w];
            apack[j] = (int)(m0 | (m1 << 8) | (m2 << 16) | (m3 << 24));
        }
        // 7-arg iu8 form: (sgn_a, A, sgn_b, B, C, reuse_a, reuse_b)
        acc = __builtin_amdgcn_wmma_i32_16x16x64_iu8(false, apack, false, ones,
                                                     acc, false, false);
    }

    unsigned local = 0;
    #pragma unroll
    for (int i = 0; i < 8; ++i) local += (unsigned)acc[i];   // 16 x partial sum

    // Scalar tail for n % 1024 (none for N=2^24). Scaled by 16 to match the
    // WMMA column-replication factor. Uses arithmetic path (no LDS dependence).
    const unsigned rem_base = nchunks << 10;
    if (wave == 0) {
        for (unsigned i = rem_base + lane; i < (unsigned)n; i += 32u)
            local += 16u * mism3((unsigned)A[i], (unsigned)B[i]);
    }

    atomicAdd(&bsum, local);            // ds_add_u32, deterministic (integer)
    __syncthreads();
    if (threadIdx.x == 0) atomicAdd(gsum, bsum);   // one global atomic per block
}

__global__ void hl_finalize(const unsigned* __restrict__ ws,
                            float* __restrict__ out, int n) {
    if (threadIdx.x == 0 && blockIdx.x == 0)
        out[0] = (float)((double)ws[0] / (16.0 * (double)n));
}

extern "C" void kernel_launch(void* const* d_in, const int* in_sizes, int n_in,
                              void* d_out, int out_size, void* d_ws, size_t ws_size,
                              hipStream_t stream) {
    const int* a = (const int*)d_in[0];     // "output" classes, [0,27)
    const int* b = (const int*)d_in[1];     // "target" classes, [0,27)
    const int  n = in_sizes[0];             // 16777216
    unsigned*  gsum = (unsigned*)d_ws;      // 4-byte integer accumulator
    float*     out  = (float*)d_out;

    hl_init_ws<<<1, 32, 0, stream>>>(gsum);
    // 1024 blocks x 256 threads = 8192 waves; each wave streams 2 chunks of 1024.
    hl_hamming_wmma<<<1024, 256, 0, stream>>>(a, b, gsum, n);
    hl_finalize<<<1, 32, 0, stream>>>(gsum, out, n);
}
